// LstmDecoder_76192719831982
// MI455X (gfx1250) — compile-verified, matrix operand order checked by reading, not run on hardware
//
#include <hip/hip_runtime.h>

// ---------------------------------------------------------------------------
// Problem dimensions (fixed by the reference)
// ---------------------------------------------------------------------------
#define T_STEPS 512
#define BATCH   512
#define FEAT    128
#define HID     512
#define TAGS    2

// LDS staging geometry: K-chunk 64, padded row stride 144B (bank-conflict-free)
#define KC      64
#define ROWB    144          // 64*2 + 16 pad
#define A_TILE  (64 * ROWB)  // 9216 B
#define B_TILE  (128 * ROWB) // 18432 B

// ---------------------------------------------------------------------------
// CDNA5 WMMA types
// ---------------------------------------------------------------------------
typedef __attribute__((ext_vector_type(16))) __bf16 v16bf;
typedef __attribute__((ext_vector_type(8)))  __bf16 v8bf;
typedef __attribute__((ext_vector_type(8)))  float  v8f;
typedef int v4i32 __attribute__((vector_size(16)));  // matches builtin's pointee type

#define GAS __attribute__((address_space(1)))
#define LAS __attribute__((address_space(3)))

#if defined(__HIP_DEVICE_COMPILE__) && __has_builtin(__builtin_amdgcn_global_load_async_to_lds_b128)
#define HAVE_ASYNC_LDS 1
#else
#define HAVE_ASYNC_LDS 0
#endif

static __device__ __forceinline__ __bf16 f2bf(float f) {
    unsigned u = __float_as_uint(f);
    unsigned r = (u + 0x7FFFu + ((u >> 16) & 1u)) >> 16;   // round-to-nearest-even
    unsigned short s = (unsigned short)r;
    __bf16 out;
    __builtin_memcpy(&out, &s, 2);
    return out;
}

static __device__ __forceinline__ float bf2f(__bf16 x) {
    unsigned short s;
    __builtin_memcpy(&s, &x, 2);
    return __uint_as_float(((unsigned)s) << 16);
}

static __device__ __forceinline__ float sigmoidf(float x) {
    return 1.0f / (1.0f + __expf(-x));
}

static __device__ __forceinline__ v8f wmma_bf16(v16bf a, v16bf b, v8f c) {
    return __builtin_amdgcn_wmma_f32_16x16x32_bf16(
        false, a, false, b, (short)0, c, false, false);
}

// ---------------------------------------------------------------------------
// Async global->LDS 16B copy (ASYNCcnt path), sync fallback via ds_store.
// ---------------------------------------------------------------------------
static __device__ __forceinline__ void cp16_to_lds(char* lds_dst, const __bf16* gsrc) {
#if HAVE_ASYNC_LDS
    __builtin_amdgcn_global_load_async_to_lds_b128(
        (GAS v4i32*)(unsigned long long)(const void*)gsrc,
        (LAS v4i32*)(unsigned)(unsigned long long)(void*)lds_dst,
        0, 0);
#else
    *(v8bf*)lds_dst = *(const v8bf*)gsrc;
#endif
}

template <int N>
static __device__ __forceinline__ void wait_async() {
#if HAVE_ASYNC_LDS
#if __has_builtin(__builtin_amdgcn_s_wait_asynccnt)
    __builtin_amdgcn_s_wait_asynccnt(N);
#else
    asm volatile("s_wait_asynccnt %0" ::"i"(N) : "memory");
#endif
#endif
}

// ---------------------------------------------------------------------------
// Fragment loaders.
// A-fragment 16x32 bf16 layout (and mirrored B for row-major W in "@ W^T"):
// lane<16 -> row=lane, K chunks {0..7,16..23}; lane>=16 -> row=lane-16,
// K chunks {8..15,24..31}: two 16-byte row segments at off and off+32.
// ---------------------------------------------------------------------------
static __device__ __forceinline__ v16bf lds_frag(const char* buf, int row, int kloc, int lane) {
    const int off = row * ROWB + (kloc + ((lane & 16) ? 8 : 0)) * 2;
    v8bf lo = *(const v8bf*)(buf + off);
    v8bf hi = *(const v8bf*)(buf + off + 32);
    v16bf r;
#pragma unroll
    for (int i = 0; i < 8; ++i) { r[i] = lo[i]; r[i + 8] = hi[i]; }
    return r;
}

static __device__ __forceinline__ v16bf load_frag_f32(const float* __restrict__ row,
                                                      int base_k, int lane) {
    const int k0 = base_k + ((lane & 16) ? 8 : 0);
    const float4 f0 = *(const float4*)(row + k0);
    const float4 f1 = *(const float4*)(row + k0 + 4);
    const float4 f2 = *(const float4*)(row + k0 + 16);
    const float4 f3 = *(const float4*)(row + k0 + 20);
    v16bf r;
    r[0]  = f2bf(f0.x); r[1]  = f2bf(f0.y); r[2]  = f2bf(f0.z); r[3]  = f2bf(f0.w);
    r[4]  = f2bf(f1.x); r[5]  = f2bf(f1.y); r[6]  = f2bf(f1.z); r[7]  = f2bf(f1.w);
    r[8]  = f2bf(f2.x); r[9]  = f2bf(f2.y); r[10] = f2bf(f2.z); r[11] = f2bf(f2.w);
    r[12] = f2bf(f3.x); r[13] = f2bf(f3.y); r[14] = f2bf(f3.z); r[15] = f2bf(f3.w);
    return r;
}

static __device__ __forceinline__ v16bf load_frag_bf16g(const __bf16* __restrict__ row,
                                                        int base_k, int lane) {
    const int k0 = base_k + ((lane & 16) ? 8 : 0);
    v8bf lo = *(const v8bf*)(row + k0);
    v8bf hi = *(const v8bf*)(row + k0 + 16);
    v16bf r;
#pragma unroll
    for (int i = 0; i < 8; ++i) { r[i] = lo[i]; r[i + 8] = hi[i]; }
    return r;
}

// ---------------------------------------------------------------------------
// Tile stagers: coalesced (8 lanes cover one 128B row segment).
// ---------------------------------------------------------------------------
static __device__ __forceinline__ void stage_A(char* buf, const __bf16* __restrict__ A,
                                               int ldA, int m_wg, int k, int tid) {
    // 64 rows x 128 B -> 512 x 16B segments, 128 threads x 4
#pragma unroll
    for (int i = 0; i < 4; ++i) {
        const int segid = tid + i * 128;
        const int row = segid >> 3, seg = segid & 7;
        cp16_to_lds(buf + row * ROWB + seg * 16,
                    A + (size_t)(m_wg + row) * ldA + k + seg * 8);
    }
}

static __device__ __forceinline__ void stage_B(char* buf, const __bf16* __restrict__ W,
                                               int ldW, int j_wg, int k, int tid) {
    // 128 rows (4 gates x 32 cols) x 128 B -> 1024 segments, 128 threads x 8
#pragma unroll
    for (int i = 0; i < 8; ++i) {
        const int segid = tid + i * 128;
        const int row = segid >> 3, seg = segid & 7;
        const int wrow = (row >> 5) * HID + j_wg + (row & 31);
        cp16_to_lds(buf + row * ROWB + seg * 16,
                    W + (size_t)wrow * ldW + k + seg * 8);
    }
}

// ---------------------------------------------------------------------------
// Fused LSTM layer step:
//   gates[B,4H] = (Af @ Wf^T)? + A1 @ W1^T + (A2 @ W2^T)? + bias
//   c = sig(f)*c + sig(i)*tanh(g);  h_out = sig(o)*tanh(c)   (h_out bf16)
// Workgroup: 4 waves, tile 64(batch) x 32(hidden); bf16 terms staged through
// double-buffered LDS via async copies; fp32 signal term direct from global.
// ---------------------------------------------------------------------------
__global__ __launch_bounds__(128)
void lstm_layer_kernel(const float*  __restrict__ Af, const __bf16* __restrict__ Wf, int Kf,
                       const __bf16* __restrict__ A1, const __bf16* __restrict__ W1, int K1,
                       const __bf16* __restrict__ A2, const __bf16* __restrict__ W2, int K2,
                       const float*  __restrict__ bias,
                       float*        __restrict__ c,
                       __bf16*       __restrict__ h_out) {
    __shared__ __align__(16) char smem[2 * A_TILE + 2 * B_TILE];  // 55296 B

    const int tid    = threadIdx.x;
    const int lane   = tid & 31;
    const int wave   = tid >> 5;
    const int m_wg   = blockIdx.x * 64;
    const int j_wg   = blockIdx.y * 32;
    const int m_base = m_wg + (wave & 1) * 32;
    const int j0     = j_wg + (wave >> 1) * 16;
    const int col    = lane & 15;
    const int hiHalf = (lane & 16) ? 1 : 0;

    v8f acc[2][4];
#pragma unroll
    for (int s = 0; s < 2; ++s)
#pragma unroll
        for (int g = 0; g < 4; ++g) {
            const float bv = bias[g * HID + j0 + col];
#pragma unroll
            for (int v = 0; v < 8; ++v) acc[s][g][v] = bv;
        }

    // ---- optional fp32-input term (signal @ W_ih1^T), direct global loads
    if (Af) {
        const float* rowA0 = Af + (size_t)(m_base +      col) * Kf;
        const float* rowA1 = Af + (size_t)(m_base + 16 + col) * Kf;
        for (int k = 0; k < Kf; k += 32) {
            v16bf a0 = load_frag_f32(rowA0, k, lane);
            v16bf a1 = load_frag_f32(rowA1, k, lane);
#pragma unroll
            for (int g = 0; g < 4; ++g) {
                v16bf b = load_frag_bf16g(Wf + (size_t)(g * HID + j0 + col) * Kf, k, lane);
                acc[0][g] = wmma_bf16(a0, b, acc[0][g]);
                acc[1][g] = wmma_bf16(a1, b, acc[1][g]);
            }
        }
    }

    // ---- LDS-staged bf16 term (double-buffered async pipeline)
    auto staged_term = [&](const __bf16* __restrict__ A, int ldA,
                           const __bf16* __restrict__ W, int ldW, int K) {
        const int stages = K / KC;
        stage_A(smem, A, ldA, m_wg, 0, tid);
        stage_B(smem + 2 * A_TILE, W, ldW, j_wg, 0, tid);
        for (int s = 0; s < stages; ++s) {
            const int cur = s & 1;
            char* curA = smem + (cur ? A_TILE : 0);
            char* curB = smem + 2 * A_TILE + (cur ? B_TILE : 0);
            if (s + 1 < stages) {
                char* nxtA = smem + (cur ? 0 : A_TILE);
                char* nxtB = smem + 2 * A_TILE + (cur ? 0 : B_TILE);
                stage_A(nxtA, A, ldA, m_wg, (s + 1) * KC, tid);
                stage_B(nxtB, W, ldW, j_wg, (s + 1) * KC, tid);
                wait_async<12>();   // current stage's 12 copies retired (in-order)
            } else {
                wait_async<0>();
            }
            __syncthreads();
#pragma unroll
            for (int kl = 0; kl < KC; kl += 32) {
                v16bf a0 = lds_frag(curA, (wave & 1) * 32 +      col, kl, lane);
                v16bf a1 = lds_frag(curA, (wave & 1) * 32 + 16 + col, kl, lane);
#pragma unroll
                for (int g = 0; g < 4; ++g) {
                    v16bf b = lds_frag(curB, g * 32 + (wave >> 1) * 16 + col, kl, lane);
                    acc[0][g] = wmma_bf16(a0, b, acc[0][g]);
                    acc[1][g] = wmma_bf16(a1, b, acc[1][g]);
                }
            }
            __syncthreads();  // protect cur buffers before they are re-staged
        }
    };

    staged_term(A1, K1, W1, K1, K1);
    if (A2) staged_term(A2, K2, W2, K2, K2);

    // ---- LSTM pointwise; C/D layout: m = v + 8*hiHalf, n = lane&15
#pragma unroll
    for (int s = 0; s < 2; ++s)
#pragma unroll
        for (int v = 0; v < 8; ++v) {
            const int brow = m_base + s * 16 + v + hiHalf * 8;
            const int idx  = brow * HID + j0 + col;
            const float iv = sigmoidf(acc[s][0][v]);
            const float fv = sigmoidf(acc[s][1][v]);
            const float gv = tanhf(acc[s][2][v]);
            const float ov = sigmoidf(acc[s][3][v]);
            const float cn = fv * c[idx] + iv * gv;
            c[idx]     = cn;
            h_out[idx] = f2bf(ov * tanhf(cn));
        }
}

// ---------------------------------------------------------------------------
// Output head: out[b, t*2+tag] = h2[b,:] . W_lin[tag,:] + b_lin[tag]
// ---------------------------------------------------------------------------
__global__ __launch_bounds__(256)
void lstm_out_kernel(const __bf16* __restrict__ h2,
                     const float*  __restrict__ W_lin,
                     const float*  __restrict__ b_lin,
                     float*        __restrict__ out, int t) {
    const int tid = blockIdx.x * blockDim.x + threadIdx.x;
    if (tid >= BATCH * TAGS) return;
    const int b = tid >> 1, tag = tid & 1;
    const __bf16* hr = h2 + (size_t)b * HID;
    const float*  wr = W_lin + (size_t)tag * HID;
    float s = b_lin[tag];
#pragma unroll 8
    for (int k = 0; k < HID; ++k) s += bf2f(hr[k]) * wr[k];
    out[(size_t)b * (T_STEPS * TAGS) + t * TAGS + tag] = s;
}

// ---------------------------------------------------------------------------
// Prep kernels (run every launch -> deterministic)
// ---------------------------------------------------------------------------
__global__ void cvt_f32_bf16_kernel(const float* __restrict__ src, __bf16* __restrict__ dst, int n) {
    const int i = blockIdx.x * blockDim.x + threadIdx.x;
    if (i < n) dst[i] = f2bf(src[i]);
}
__global__ void bias_sum_kernel(const float* __restrict__ a, const float* __restrict__ b,
                                float* __restrict__ dst, int n) {
    const int i = blockIdx.x * blockDim.x + threadIdx.x;
    if (i < n) dst[i] = a[i] + b[i];
}
__global__ void zero_u32_kernel(unsigned* __restrict__ p, int nwords) {
    const int i = blockIdx.x * blockDim.x + threadIdx.x;
    if (i < nwords) p[i] = 0u;
}

// ---------------------------------------------------------------------------
// Host driver — three kernels per timestep recorded on `stream`; graph capture
// turns the T=512 sequential dependence into cheap graph-node replays.
// ---------------------------------------------------------------------------
extern "C" void kernel_launch(void* const* d_in, const int* in_sizes, int n_in,
                              void* d_out, int out_size, void* d_ws, size_t ws_size,
                              hipStream_t stream) {
    (void)in_sizes; (void)n_in; (void)out_size; (void)ws_size;

    const float* signal = (const float*)d_in[0];   // [T,B,F]
    const float* W_ih1  = (const float*)d_in[1];   // [4H,F]
    const float* W_hh1  = (const float*)d_in[2];   // [4H,H]
    const float* b_ih1  = (const float*)d_in[3];
    const float* b_hh1  = (const float*)d_in[4];
    const float* W_ih2  = (const float*)d_in[5];   // [4H,H]
    const float* W_hh2  = (const float*)d_in[6];   // [4H,H]
    const float* b_ih2  = (const float*)d_in[7];
    const float* b_hh2  = (const float*)d_in[8];
    const float* W_lin  = (const float*)d_in[9];   // [TAG,H]
    const float* b_lin  = (const float*)d_in[10];  // [TAG]
    float* out = (float*)d_out;

    // ---- workspace layout (~10.6 MB), 256B aligned slots
    char* base = (char*)d_ws;
    size_t off = 0;
    auto take = [&](size_t bytes) -> char* {
        char* p = base + off;
        off += (bytes + 255) & ~(size_t)255;
        return p;
    };
    __bf16* Wih1b = (__bf16*)take((size_t)4 * HID * FEAT * 2);
    __bf16* Whh1b = (__bf16*)take((size_t)4 * HID * HID * 2);
    __bf16* Wih2b = (__bf16*)take((size_t)4 * HID * HID * 2);
    __bf16* Whh2b = (__bf16*)take((size_t)4 * HID * HID * 2);
    float*  bias1 = (float*)take((size_t)4 * HID * 4);
    float*  bias2 = (float*)take((size_t)4 * HID * 4);
    __bf16* h1buf = (__bf16*)take((size_t)2 * BATCH * HID * 2);  // double buffered
    __bf16* h2buf = (__bf16*)take((size_t)2 * BATCH * HID * 2);  // double buffered
    float*  c1    = (float*)take((size_t)BATCH * HID * 4);
    float*  c2    = (float*)take((size_t)BATCH * HID * 4);

    // ---- prep: bf16 weights, fused biases, zero state
    int n;
    n = 4 * HID * FEAT;
    cvt_f32_bf16_kernel<<<(n + 255) / 256, 256, 0, stream>>>(W_ih1, Wih1b, n);
    n = 4 * HID * HID;
    cvt_f32_bf16_kernel<<<(n + 255) / 256, 256, 0, stream>>>(W_hh1, Whh1b, n);
    cvt_f32_bf16_kernel<<<(n + 255) / 256, 256, 0, stream>>>(W_ih2, Wih2b, n);
    cvt_f32_bf16_kernel<<<(n + 255) / 256, 256, 0, stream>>>(W_hh2, Whh2b, n);
    n = 4 * HID;
    bias_sum_kernel<<<(n + 255) / 256, 256, 0, stream>>>(b_ih1, b_hh1, bias1, n);
    bias_sum_kernel<<<(n + 255) / 256, 256, 0, stream>>>(b_ih2, b_hh2, bias2, n);
    n = (2 * BATCH * HID * 2) / 4;  // both h buffers, as u32 words
    zero_u32_kernel<<<(n + 255) / 256, 256, 0, stream>>>((unsigned*)h1buf, n);
    zero_u32_kernel<<<(n + 255) / 256, 256, 0, stream>>>((unsigned*)h2buf, n);
    n = BATCH * HID;
    zero_u32_kernel<<<(n + 255) / 256, 256, 0, stream>>>((unsigned*)c1, n);
    zero_u32_kernel<<<(n + 255) / 256, 256, 0, stream>>>((unsigned*)c2, n);

    // ---- sequential scan: stream ordering provides the step barrier
    const dim3 lgrid(BATCH / 64, HID / 32);  // 8 x 16 = 128 workgroups
    for (int t = 0; t < T_STEPS; ++t) {
        const int cur = t & 1, nxt = cur ^ 1;
        const float* x_t = signal + (size_t)t * BATCH * FEAT;
        __bf16* h1_cur = h1buf + (size_t)cur * BATCH * HID;
        __bf16* h1_nxt = h1buf + (size_t)nxt * BATCH * HID;
        __bf16* h2_cur = h2buf + (size_t)cur * BATCH * HID;
        __bf16* h2_nxt = h2buf + (size_t)nxt * BATCH * HID;

        // Layer 1: gates = x_t @ W_ih1^T + h1 @ W_hh1^T + bias1
        lstm_layer_kernel<<<lgrid, 128, 0, stream>>>(
            x_t, Wih1b, FEAT,
            h1_cur, Whh1b, HID,
            nullptr, nullptr, 0,
            bias1, c1, h1_nxt);

        // Layer 2: gates = h1' @ W_ih2^T + h2 @ W_hh2^T + bias2
        lstm_layer_kernel<<<lgrid, 128, 0, stream>>>(
            nullptr, nullptr, 0,
            h1_nxt, Wih2b, HID,
            h2_cur, Whh2b, HID,
            bias2, c2, h2_nxt);

        // Output head
        lstm_out_kernel<<<(BATCH * TAGS + 255) / 256, 256, 0, stream>>>(
            h2_nxt, W_lin, b_lin, out, t);
    }
}
